// MinGRUBlock_80530636800512
// MI455X (gfx1250) — matrix operand order repaired and checked by reading, not compiled, not run on hardware
//
#include <hip/hip_runtime.h>
#include <stdint.h>

// Problem sizes (fixed by the reference)
#define B_    8
#define S_    2048
#define D_    1024
#define F_    3072
#define NTOK  (B_ * S_)     // 16384 tokens
#define NC    16            // scan chunks
#define CH    (S_ / NC)     // 128 steps per chunk

typedef __attribute__((ext_vector_type(16))) __bf16 v16bf;
typedef __attribute__((ext_vector_type(8)))  float  v8f;

union FragBF { v16bf v; uint4 q[2]; };

__device__ __forceinline__ unsigned short f2bf(float f) {
  union { float f; unsigned int u; } c; c.f = f;
  unsigned int u = c.u;
  unsigned int r = (u + 0x7FFFu + ((u >> 16) & 1u)) >> 16;  // round-to-nearest-even
  return (unsigned short)r;
}

__device__ __forceinline__ v8f vzero8() {
  v8f z;
#pragma unroll
  for (int i = 0; i < 8; ++i) z[i] = 0.f;
  return z;
}

__device__ __forceinline__ float sigmoidf_(float x) { return 1.f / (1.f + __expf(-x)); }

// ---- CDNA5 async global->LDS copy (16B per lane), tracked by ASYNCcnt ----
// LDS aperture decode takes addr[31:0] as the wave-relative LDS offset, so the
// low 32 bits of a generic pointer into __shared__ give the VDST LDS address.
__device__ __forceinline__ void async_cp16(const unsigned short* g, unsigned short* l) {
  uint32_t loff = (uint32_t)(uintptr_t)l;
  asm volatile("global_load_async_to_lds_b128 %0, %1, off"
               :
               : "v"(loff), "v"(g)
               : "memory");
}

// Async loads complete in order, so waiting ASYNCcnt <= N leaves only the
// newest N (the most recently issued tile) possibly in flight.
template <int N>
__device__ __forceinline__ void wait_async() {
#if __has_builtin(__builtin_amdgcn_s_wait_asynccnt)
  __builtin_amdgcn_s_wait_asynccnt(N);
#else
  asm volatile("s_wait_asynccnt %0" ::"i"(N) : "memory");
#endif
}

// ---- WMMA fragment loads from K-major LDS tiles (bf16) ----
// A (16x32, MxK): lanes 0-15 rows M, elems 0..15 = K(koff+0..15),
// koff = 0 (lanes 0-15) / 16 (lanes 16-31).
__device__ __forceinline__ v16bf load_a_frag(const unsigned short* As, int lane, int mbase) {
  FragBF f;
  const int m    = mbase + (lane & 15);
  const int koff = (lane & 16) ? 16 : 0;
  const unsigned short* p = As + m * 32;
  f.q[0] = *reinterpret_cast<const uint4*>(p + koff);
  f.q[1] = *reinterpret_cast<const uint4*>(p + 8 + koff);
  return f.v;
}
// B (32x16, KxN): lanes 0-15 cols N with K=0..15, lanes 16-31 same cols with K=16..31.
__device__ __forceinline__ v16bf load_b_frag(const unsigned short* Bs, int lane, int nbase) {
  FragBF f;
  const int n    = nbase + (lane & 15);
  const int koff = (lane & 16) ? 16 : 0;
  const unsigned short* p = Bs + n * 32 + koff;
  f.q[0] = *reinterpret_cast<const uint4*>(p);
  f.q[1] = *reinterpret_cast<const uint4*>(p + 8);
  return f.v;
}

// ---------------- weight f32 -> bf16 ----------------
__global__ void cvt_f32_bf16(const float* __restrict__ src, unsigned short* __restrict__ dst, int n) {
  int i = blockIdx.x * 256 + threadIdx.x;
  if (i < n) dst[i] = f2bf(src[i]);
}

// ---------------- RMSNorm (fp32 in, bf16 out) ----------------
__global__ __launch_bounds__(256) void rmsnorm_bf16(const float* __restrict__ x,
                                                    const float* __restrict__ w,
                                                    unsigned short* __restrict__ out) {
  __shared__ float red[256];
  const int row = blockIdx.x;
  const int tid = threadIdx.x;
  const float* xr = x + (size_t)row * D_;
  float vals[4];
  float s = 0.f;
#pragma unroll
  for (int i = 0; i < 4; ++i) { float v = xr[tid + i * 256]; vals[i] = v; s += v * v; }
  red[tid] = s;
  __syncthreads();
  for (int off = 128; off > 0; off >>= 1) {
    if (tid < off) red[tid] += red[tid + off];
    __syncthreads();
  }
  const float rms = rsqrtf(red[0] * (1.0f / D_) + 1e-6f);
  unsigned short* orow = out + (size_t)row * D_;
#pragma unroll
  for (int i = 0; i < 4; ++i) { int c = tid + i * 256; orow[c] = f2bf(vals[i] * rms * w[c]); }
}

// ---------------- fused g/v/d GEMM + gate epilogue ----------------
// Block tile 128x64, 8 waves (4x2), wave tile 32x32 (2x2 WMMA frags), BK=32.
// 3-stage async pipeline, one barrier per K-step, branch-free steady state.
__global__ __launch_bounds__(256) void gemm_gvd(const unsigned short* __restrict__ xnb,
                                                const unsigned short* __restrict__ wgb,
                                                const unsigned short* __restrict__ wvb,
                                                const unsigned short* __restrict__ wdb,
                                                const float* __restrict__ bg,
                                                const float* __restrict__ bv,
                                                const float* __restrict__ bd,
                                                float* __restrict__ xs,
                                                float* __restrict__ ab) {
  __shared__ __align__(16) unsigned short As[3][128 * 32];
  __shared__ __align__(16) unsigned short Bs[3][3][64 * 32];
  const int tid = threadIdx.x;
  const int lane = tid & 31;
  const int wave = tid >> 5;
  const int wm = wave >> 1, wn = wave & 1;
  const int rowBase = blockIdx.x * 128;
  const int colBase = blockIdx.y * 64;
  const unsigned short* wptr[3] = { wgb, wvb, wdb };

  v8f acc[3][2][2];
#pragma unroll
  for (int w = 0; w < 3; ++w)
#pragma unroll
    for (int i = 0; i < 2; ++i)
#pragma unroll
      for (int j = 0; j < 2; ++j) acc[w][i][j] = vzero8();

  const int ar = tid >> 1, aseg = (tid & 1) * 16;
  const int br = tid >> 2, bseg = (tid & 3) * 8;
  const int NISSUE = 5;  // asyncs per thread per tile

  auto issue_tiles = [&](int buf, int k0) {
    const unsigned short* ga = xnb + (size_t)(rowBase + ar) * D_ + k0 + aseg;
    async_cp16(ga,     &As[buf][ar * 32 + aseg]);
    async_cp16(ga + 8, &As[buf][ar * 32 + aseg + 8]);
#pragma unroll
    for (int w = 0; w < 3; ++w)
      async_cp16(wptr[w] + (size_t)(colBase + br) * D_ + k0 + bseg, &Bs[buf][w][br * 32 + bseg]);
  };

  auto compute = [&](int buf) {
    v16bf af0 = load_a_frag(As[buf], lane, wm * 32 + 0);
    v16bf af1 = load_a_frag(As[buf], lane, wm * 32 + 16);
#pragma unroll
    for (int w = 0; w < 3; ++w) {
#pragma unroll
      for (int tn = 0; tn < 2; ++tn) {
        v16bf bf = load_b_frag(&Bs[buf][w][0], lane, wn * 32 + tn * 16);
        acc[w][0][tn] = __builtin_amdgcn_wmma_f32_16x16x32_bf16(false, af0, false, bf, (short)0,
                                                                acc[w][0][tn], false, false);
        acc[w][1][tn] = __builtin_amdgcn_wmma_f32_16x16x32_bf16(false, af1, false, bf, (short)0,
                                                                acc[w][1][tn], false, false);
      }
    }
  };

  issue_tiles(0, 0);
  issue_tiles(1, 32);
  int buf = 0;
  for (int k0 = 0; k0 < D_ - 64; k0 += 32) {   // k = 0 .. nk-3: branch-free steady state
    wait_async<NISSUE>();
    __syncthreads();
    issue_tiles(buf == 0 ? 2 : buf - 1, k0 + 64);
    compute(buf);
    buf = (buf == 2) ? 0 : buf + 1;
  }
  wait_async<NISSUE>();                        // k = nk-2: nothing new to issue
  __syncthreads();
  compute(buf);
  buf = (buf == 2) ? 0 : buf + 1;
  wait_async<0>();                             // k = nk-1: drain
  __syncthreads();
  compute(buf);

  // Epilogue: x_scan = sigmoid(g)*tanh(v); a = 0.001 + 0.998*sigmoid(d)
  const int ln = lane & 15;
  const int mh = (lane >> 4) << 3;
#pragma unroll
  for (int tm = 0; tm < 2; ++tm)
#pragma unroll
    for (int tn = 0; tn < 2; ++tn) {
      const int col = colBase + wn * 32 + tn * 16 + ln;
      const float bgv = bg[col], bvv = bv[col], bdv = bd[col];
#pragma unroll
      for (int r = 0; r < 8; ++r) {
        const int row = rowBase + wm * 32 + tm * 16 + mh + r;
        const size_t idx = (size_t)row * D_ + col;
        const float g = acc[0][tm][tn][r] + bgv;
        const float v = acc[1][tm][tn][r] + bvv;
        const float d = acc[2][tm][tn][r] + bdv;
        xs[idx] = sigmoidf_(g) * tanhf(v);
        ab[idx] = 0.001f + 0.998f * sigmoidf_(d);
      }
    }
}

// ---------------- chunked causal scan ----------------
__global__ __launch_bounds__(256) void scan_pass1(float* __restrict__ xs,
                                                  const float* __restrict__ ab,
                                                  float* __restrict__ hlast,
                                                  float* __restrict__ pprod) {
  const int gid = blockIdx.x * 256 + threadIdx.x;   // B*D*NC
  const int d = gid % D_;
  const int rest = gid / D_;
  const int c = rest % NC;
  const int b = rest / NC;
  const size_t base = ((size_t)b * S_ + (size_t)c * CH) * D_ + d;
  float h = 0.f, p = 1.f;
  for (int t = 0; t < CH; ++t) {
    const size_t idx = base + (size_t)t * D_;
    const float a = ab[idx];
    h = a * h + xs[idx];
    p *= a;
    xs[idx] = h;
  }
  const int ci = (b * D_ + d) * NC + c;
  hlast[ci] = h;
  pprod[ci] = p;
}

__global__ __launch_bounds__(256) void scan_pass2(const float* __restrict__ hlast,
                                                  const float* __restrict__ pprod,
                                                  float* __restrict__ carryIn) {
  const int gid = blockIdx.x * 256 + threadIdx.x;   // B*D
  const float* hl = hlast + (size_t)gid * NC;
  const float* pp = pprod + (size_t)gid * NC;
  float* ci = carryIn + (size_t)gid * NC;
  float H = 0.f;
  for (int c = 0; c < NC; ++c) {
    ci[c] = H;
    H = hl[c] + pp[c] * H;
  }
}

__global__ __launch_bounds__(256) void scan_pass3(const float* __restrict__ hloc,
                                                  const float* __restrict__ ab,
                                                  const float* __restrict__ carryIn,
                                                  const float* __restrict__ x,
                                                  float* __restrict__ x1) {
  const int gid = blockIdx.x * 256 + threadIdx.x;   // B*D*NC
  const int d = gid % D_;
  const int rest = gid / D_;
  const int c = rest % NC;
  const int b = rest / NC;
  const size_t base = ((size_t)b * S_ + (size_t)c * CH) * D_ + d;
  const float cin = carryIn[(b * D_ + d) * NC + c];
  float p = 1.f;
  for (int t = 0; t < CH; ++t) {
    const size_t idx = base + (size_t)t * D_;
    p *= ab[idx];
    const float h = hloc[idx] + p * cin;
    x1[idx] = x[idx] + h;
  }
}

// ---------------- fused gate/up GEMM + SiLU epilogue (bf16 out) ----------------
__global__ __launch_bounds__(256) void gemm_gateup(const unsigned short* __restrict__ xnb,
                                                   const unsigned short* __restrict__ wgateb,
                                                   const unsigned short* __restrict__ wupb,
                                                   unsigned short* __restrict__ hb) {
  __shared__ __align__(16) unsigned short As[3][128 * 32];
  __shared__ __align__(16) unsigned short Bs[3][2][64 * 32];
  const int tid = threadIdx.x;
  const int lane = tid & 31;
  const int wave = tid >> 5;
  const int wm = wave >> 1, wn = wave & 1;
  const int rowBase = blockIdx.x * 128;
  const int colBase = blockIdx.y * 64;
  const unsigned short* wptr[2] = { wgateb, wupb };

  v8f acc[2][2][2];
#pragma unroll
  for (int w = 0; w < 2; ++w)
#pragma unroll
    for (int i = 0; i < 2; ++i)
#pragma unroll
      for (int j = 0; j < 2; ++j) acc[w][i][j] = vzero8();

  const int ar = tid >> 1, aseg = (tid & 1) * 16;
  const int br = tid >> 2, bseg = (tid & 3) * 8;
  const int NISSUE = 4;

  auto issue_tiles = [&](int buf, int k0) {
    const unsigned short* ga = xnb + (size_t)(rowBase + ar) * D_ + k0 + aseg;
    async_cp16(ga,     &As[buf][ar * 32 + aseg]);
    async_cp16(ga + 8, &As[buf][ar * 32 + aseg + 8]);
#pragma unroll
    for (int w = 0; w < 2; ++w)
      async_cp16(wptr[w] + (size_t)(colBase + br) * D_ + k0 + bseg, &Bs[buf][w][br * 32 + bseg]);
  };

  auto compute = [&](int buf) {
    v16bf af0 = load_a_frag(As[buf], lane, wm * 32 + 0);
    v16bf af1 = load_a_frag(As[buf], lane, wm * 32 + 16);
#pragma unroll
    for (int w = 0; w < 2; ++w) {
#pragma unroll
      for (int tn = 0; tn < 2; ++tn) {
        v16bf bf = load_b_frag(&Bs[buf][w][0], lane, wn * 32 + tn * 16);
        acc[w][0][tn] = __builtin_amdgcn_wmma_f32_16x16x32_bf16(false, af0, false, bf, (short)0,
                                                                acc[w][0][tn], false, false);
        acc[w][1][tn] = __builtin_amdgcn_wmma_f32_16x16x32_bf16(false, af1, false, bf, (short)0,
                                                                acc[w][1][tn], false, false);
      }
    }
  };

  issue_tiles(0, 0);
  issue_tiles(1, 32);
  int buf = 0;
  for (int k0 = 0; k0 < D_ - 64; k0 += 32) {
    wait_async<NISSUE>();
    __syncthreads();
    issue_tiles(buf == 0 ? 2 : buf - 1, k0 + 64);
    compute(buf);
    buf = (buf == 2) ? 0 : buf + 1;
  }
  wait_async<NISSUE>();
  __syncthreads();
  compute(buf);
  buf = (buf == 2) ? 0 : buf + 1;
  wait_async<0>();
  __syncthreads();
  compute(buf);

  const int ln = lane & 15;
  const int mh = (lane >> 4) << 3;
#pragma unroll
  for (int tm = 0; tm < 2; ++tm)
#pragma unroll
    for (int tn = 0; tn < 2; ++tn) {
      const int col = colBase + wn * 32 + tn * 16 + ln;
#pragma unroll
      for (int r = 0; r < 8; ++r) {
        const int row = rowBase + wm * 32 + tm * 16 + mh + r;
        const float g = acc[0][tm][tn][r];
        const float u = acc[1][tm][tn][r];
        hb[(size_t)row * F_ + col] = f2bf(g * sigmoidf_(g) * u);  // silu(g)*u
      }
    }
}

// ---------------- output GEMM (K=F) + residual epilogue ----------------
__global__ __launch_bounds__(256) void gemm_out(const unsigned short* __restrict__ hb,
                                                const unsigned short* __restrict__ woutb,
                                                const float* __restrict__ x1,
                                                float* __restrict__ out) {
  __shared__ __align__(16) unsigned short As[3][128 * 32];
  __shared__ __align__(16) unsigned short Bs[3][64 * 32];
  const int tid = threadIdx.x;
  const int lane = tid & 31;
  const int wave = tid >> 5;
  const int wm = wave >> 1, wn = wave & 1;
  const int rowBase = blockIdx.x * 128;
  const int colBase = blockIdx.y * 64;

  v8f acc[2][2];
#pragma unroll
  for (int i = 0; i < 2; ++i)
#pragma unroll
    for (int j = 0; j < 2; ++j) acc[i][j] = vzero8();

  const int ar = tid >> 1, aseg = (tid & 1) * 16;
  const int br = tid >> 2, bseg = (tid & 3) * 8;
  const int NISSUE = 3;

  auto issue_tiles = [&](int buf, int k0) {
    const unsigned short* ga = hb + (size_t)(rowBase + ar) * F_ + k0 + aseg;
    async_cp16(ga,     &As[buf][ar * 32 + aseg]);
    async_cp16(ga + 8, &As[buf][ar * 32 + aseg + 8]);
    async_cp16(woutb + (size_t)(colBase + br) * F_ + k0 + bseg, &Bs[buf][br * 32 + bseg]);
  };

  auto compute = [&](int buf) {
    v16bf af0 = load_a_frag(As[buf], lane, wm * 32 + 0);
    v16bf af1 = load_a_frag(As[buf], lane, wm * 32 + 16);
#pragma unroll
    for (int tn = 0; tn < 2; ++tn) {
      v16bf bf = load_b_frag(Bs[buf], lane, wn * 32 + tn * 16);
      acc[0][tn] = __builtin_amdgcn_wmma_f32_16x16x32_bf16(false, af0, false, bf, (short)0,
                                                           acc[0][tn], false, false);
      acc[1][tn] = __builtin_amdgcn_wmma_f32_16x16x32_bf16(false, af1, false, bf, (short)0,
                                                           acc[1][tn], false, false);
    }
  };

  issue_tiles(0, 0);
  issue_tiles(1, 32);
  int buf = 0;
  for (int k0 = 0; k0 < F_ - 64; k0 += 32) {
    wait_async<NISSUE>();
    __syncthreads();
    issue_tiles(buf == 0 ? 2 : buf - 1, k0 + 64);
    compute(buf);
    buf = (buf == 2) ? 0 : buf + 1;
  }
  wait_async<NISSUE>();
  __syncthreads();
  compute(buf);
  buf = (buf == 2) ? 0 : buf + 1;
  wait_async<0>();
  __syncthreads();
  compute(buf);

  const int ln = lane & 15;
  const int mh = (lane >> 4) << 3;
#pragma unroll
  for (int tm = 0; tm < 2; ++tm)
#pragma unroll
    for (int tn = 0; tn < 2; ++tn) {
      const int col = colBase + wn * 32 + tn * 16 + ln;
#pragma unroll
      for (int r = 0; r < 8; ++r) {
        const int row = rowBase + wm * 32 + tm * 16 + mh + r;
        const size_t idx = (size_t)row * D_ + col;
        out[idx] = x1[idx] + acc[tm][tn][r];
      }
    }
}

// ---------------- launcher ----------------
extern "C" void kernel_launch(void* const* d_in, const int* in_sizes, int n_in,
                              void* d_out, int out_size, void* d_ws, size_t ws_size,
                              hipStream_t stream) {
  (void)in_sizes; (void)n_in; (void)out_size; (void)ws_size;
  const float* x     = (const float*)d_in[0];
  const float* rmsmw = (const float*)d_in[1];
  const float* rmsfw = (const float*)d_in[2];
  const float* Wg    = (const float*)d_in[3];
  const float* bg    = (const float*)d_in[4];
  const float* Wv    = (const float*)d_in[5];
  const float* bv    = (const float*)d_in[6];
  const float* Wd    = (const float*)d_in[7];
  const float* bd    = (const float*)d_in[8];
  const float* Wgate = (const float*)d_in[9];
  const float* Wup   = (const float*)d_in[10];
  const float* Wout  = (const float*)d_in[11];
  float* out = (float*)d_out;

  char* ws = (char*)d_ws;
  size_t off = 0;
  auto alloc = [&](size_t bytes) { size_t o = off; off += (bytes + 255) & ~(size_t)255; return o; };

  const size_t nW = (size_t)3 * D_ * D_ + (size_t)3 * F_ * D_;   // all 6 weight mats, bf16
  const size_t o_wbf = alloc(nW * 2);
  const size_t o_xnb = alloc((size_t)NTOK * D_ * 2);
  const size_t o_xs  = alloc((size_t)NTOK * D_ * 4);
  const size_t o_ab  = alloc((size_t)NTOK * D_ * 4);
  const size_t o_x1  = alloc((size_t)NTOK * D_ * 4);
  const size_t o_hl  = alloc((size_t)B_ * D_ * NC * 4);
  const size_t o_pp  = alloc((size_t)B_ * D_ * NC * 4);
  const size_t o_ci  = alloc((size_t)B_ * D_ * NC * 4);

  unsigned short* wb = (unsigned short*)(ws + o_wbf);
  unsigned short* wgb    = wb;
  unsigned short* wvb    = wb + (size_t)D_ * D_;
  unsigned short* wdb    = wb + (size_t)2 * D_ * D_;
  unsigned short* wgateb = wb + (size_t)3 * D_ * D_;
  unsigned short* wupb   = wgateb + (size_t)F_ * D_;
  unsigned short* woutb  = wupb + (size_t)F_ * D_;

  unsigned short* xnb = (unsigned short*)(ws + o_xnb);
  float* xs = (float*)(ws + o_xs);
  float* ab = (float*)(ws + o_ab);
  float* x1 = (float*)(ws + o_x1);
  float* hl = (float*)(ws + o_hl);
  float* pp = (float*)(ws + o_pp);
  float* ci = (float*)(ws + o_ci);
  // hb (N x F bf16, 100 MB) aliases the xs+ab region (134 MB), dead after the scan.
  unsigned short* hb = (unsigned short*)(ws + o_xs);

  const int DD = D_ * D_, FD = F_ * D_;
  cvt_f32_bf16<<<(DD + 255) / 256, 256, 0, stream>>>(Wg, wgb, DD);
  cvt_f32_bf16<<<(DD + 255) / 256, 256, 0, stream>>>(Wv, wvb, DD);
  cvt_f32_bf16<<<(DD + 255) / 256, 256, 0, stream>>>(Wd, wdb, DD);
  cvt_f32_bf16<<<(FD + 255) / 256, 256, 0, stream>>>(Wgate, wgateb, FD);
  cvt_f32_bf16<<<(FD + 255) / 256, 256, 0, stream>>>(Wup, wupb, FD);
  cvt_f32_bf16<<<(FD + 255) / 256, 256, 0, stream>>>(Wout, woutb, FD);

  rmsnorm_bf16<<<NTOK, 256, 0, stream>>>(x, rmsmw, xnb);

  gemm_gvd<<<dim3(NTOK / 128, D_ / 64), 256, 0, stream>>>(xnb, wgb, wvb, wdb, bg, bv, bd, xs, ab);

  scan_pass1<<<(B_ * D_ * NC) / 256, 256, 0, stream>>>(xs, ab, hl, pp);
  scan_pass2<<<(B_ * D_) / 256, 256, 0, stream>>>(hl, pp, ci);
  scan_pass3<<<(B_ * D_ * NC) / 256, 256, 0, stream>>>(xs, ab, ci, x, x1);

  rmsnorm_bf16<<<NTOK, 256, 0, stream>>>(x1, rmsfw, xnb);

  gemm_gateup<<<dim3(NTOK / 128, F_ / 64), 256, 0, stream>>>(xnb, wgateb, wupb, hb);

  gemm_out<<<dim3(NTOK / 128, D_ / 64), 256, 0, stream>>>(hb, woutb, x1, out);
}
